// Attention_52836687675905
// MI455X (gfx1250) — matrix operand order repaired and checked by reading, not compiled, not run on hardware
//
#include <hip/hip_runtime.h>

typedef __bf16 v16bf __attribute__((ext_vector_type(16)));
typedef __bf16 v8bf  __attribute__((ext_vector_type(8)));
typedef __bf16 v4bf  __attribute__((ext_vector_type(4)));
typedef float  v8f   __attribute__((ext_vector_type(8)));

#define H_   16
#define D_   128
#define B_   2
#define S_   2048
#define DIM  2048

// ---------------------------------------------------------------------------
// CDNA5 TDM path (TENSORcnt): one tensor_load_to_lds per tile instead of
// 2048 per-lane loads. D# layout per CDNA5 ISA.
// Toolchain uses the 6-arg form: (v4u g0, v8i g1, v4i, v4i, v8i, i32 cpol).
// ---------------------------------------------------------------------------
#if __has_builtin(__builtin_amdgcn_tensor_load_to_lds)
#define HAS_TDM 1
typedef unsigned int v4u_g __attribute__((vector_size(16)));
typedef int          v8i_g __attribute__((vector_size(32)));
typedef int          v4i_t __attribute__((vector_size(16)));

// 2D tile load: tile_d0 x tile_d1 elements (bf16) from row-major tensor with
// row stride stride0 (elements) into contiguous LDS at lds_off.
__device__ __forceinline__ void tdm_load_2d(const __bf16* gtile, unsigned lds_off,
                                            unsigned tensor_d0, unsigned tensor_d1,
                                            unsigned tile_d0, unsigned tile_d1,
                                            unsigned stride0) {
  unsigned long long ga = (unsigned long long)gtile;
  v4u_g g0;
  g0[0] = 1u;                                            // count=1, user D#
  g0[1] = lds_off;                                       // LDS byte address
  g0[2] = (unsigned)(ga & 0xffffffffu);                  // global_addr[31:0]
  g0[3] = ((unsigned)((ga >> 32) & 0x01ffffffu)) | 0x80000000u;  // [56:32] | type=2
  v8i_g g1;
  g1[0] = (int)(1u << 16);                               // data_size=1 (2 bytes)
  g1[1] = (int)((tensor_d0 & 0xffffu) << 16);            // tensor_dim0[15:0]
  g1[2] = (int)((tensor_d0 >> 16) | ((tensor_d1 & 0xffffu) << 16));
  g1[3] = (int)((tensor_d1 >> 16) | (tile_d0 << 16));    // tile_dim0
  g1[4] = (int)tile_d1;                                  // tile_dim1 (tile_dim2=0)
  g1[5] = (int)stride0;                                  // tensor_dim0_stride[31:0]
  g1[6] = 0;
  g1[7] = 0;
  v4i_t z4 = {0, 0, 0, 0};                               // 2D: groups 2/3 unused
  v8i_g z8 = {0, 0, 0, 0, 0, 0, 0, 0};
  __builtin_amdgcn_tensor_load_to_lds(g0, g1, z4, z4, z8, 0);
}

__device__ __forceinline__ unsigned lds_off_of(const void* p) {
  return (unsigned)(unsigned long long)p;                // low 32 bits = LDS addr
}
#endif

__device__ __forceinline__ void wait_tensor0() {
#if __has_builtin(__builtin_amdgcn_s_wait_tensorcnt)
  __builtin_amdgcn_s_wait_tensorcnt(0);
  asm volatile("" ::: "memory");
#else
  asm volatile("s_wait_tensorcnt 0" ::: "memory");
#endif
}

// ---------------------------------------------------------------------------
// CDNA5 async global->LDS path (ASYNCcnt), toolchain-confirmed prototype.
// ---------------------------------------------------------------------------
#if __has_builtin(__builtin_amdgcn_global_load_async_to_lds_b128)
#define HAS_ASYNC_LDS 1
typedef int v4i_g __attribute__((vector_size(16)));
typedef __attribute__((address_space(1))) v4i_g* as1_int4p;
typedef __attribute__((address_space(3))) v4i_g* as3_int4p;

__device__ __forceinline__ void async_cp16(const __bf16* g, __bf16* l) {
  __builtin_amdgcn_global_load_async_to_lds_b128(
      (as1_int4p)(unsigned long long)g,
      (as3_int4p)(unsigned int)(unsigned long long)l,
      0, 0);
}
#endif

__device__ __forceinline__ void wait_async0() {
#if __has_builtin(__builtin_amdgcn_s_wait_asynccnt)
  __builtin_amdgcn_s_wait_asynccnt(0);
  asm volatile("" ::: "memory");
#else
  asm volatile("s_wait_asynccnt 0" ::: "memory");
#endif
}

__device__ __forceinline__ v8f wmma_bf16(v16bf a, v16bf b, v8f c) {
  return __builtin_amdgcn_wmma_f32_16x16x32_bf16(false, a, false, b, (short)0, c,
                                                 false, false);
}

__device__ __forceinline__ v16bf ld_frag(const __bf16* p0, const __bf16* p1) {
  v8bf lo = *(const v8bf*)p0;
  v8bf hi = *(const v8bf*)p1;
  v16bf r;
#pragma unroll
  for (int i = 0; i < 8; ++i) { r[i] = lo[i]; r[i + 8] = hi[i]; }
  return r;
}

// ---------------------------------------------------------------------------
// fp32 -> bf16 conversion (vectorized x4)
// ---------------------------------------------------------------------------
__global__ void cvt_f32_bf16(const float* __restrict__ in,
                             __bf16* __restrict__ out, int n4) {
  int i = blockIdx.x * blockDim.x + threadIdx.x;
  if (i < n4) {
    float4 f = ((const float4*)in)[i];
    v4bf o;
    o[0] = (__bf16)f.x; o[1] = (__bf16)f.y; o[2] = (__bf16)f.z; o[3] = (__bf16)f.w;
    ((v4bf*)out)[i] = o;
  }
}

// ---------------------------------------------------------------------------
// Shared GEMM core: C(128x128) += A(128xK) * W(NxK)^T, K-tile 64,
// 8 waves, wave tile 32x64 (2x4 WMMA tiles), LDS double buffer.
// Tile staging priority: TDM > async-to-LDS > register double buffer.
// ---------------------------------------------------------------------------
__device__ __forceinline__ void gemm_core(const __bf16* __restrict__ A,
                                          const __bf16* __restrict__ W,
                                          int m0, int n0, int ntot,
                                          v8f acc[2][4]) {
  const int tid  = threadIdx.x;
  const int wave = tid >> 5, lane = tid & 31;
  const int hh   = lane >> 4, ln = lane & 15;
  const int wm   = (wave & 3) * 32;
  const int wn   = (wave >> 2) * 64;
  const int lr   = tid >> 3;          // 0..31: row base within 32-row group
  const int lc   = (tid & 7) * 8;     // 0..56: element col of 16B chunk
  (void)lr; (void)lc; (void)ntot;

#if defined(HAS_TDM)
  __shared__ __align__(16) __bf16 As[2][128][64];   // 2 x 16 KB
  __shared__ __align__(16) __bf16 Bs[2][128][64];   // 2 x 16 KB

  const __bf16* Atile = A + (size_t)m0 * DIM;       // 128 x DIM row block
  const __bf16* Wtile = W + (size_t)n0 * DIM;
  if (wave == 0) {                    // one wave drives the DMA engine
    tdm_load_2d(Atile, lds_off_of(&As[0][0][0]), DIM, 4096, 64, 128, DIM);
    tdm_load_2d(Wtile, lds_off_of(&Bs[0][0][0]), DIM, (unsigned)ntot, 64, 128, DIM);
    wait_tensor0();
  }
  __syncthreads();

  int cur = 0;
  for (int k0 = 0; k0 < DIM; k0 += 64) {
    const bool more = (k0 + 64) < DIM;
    if (more && wave == 0) {          // DMA next tile into other buffer
      tdm_load_2d(Atile + k0 + 64, lds_off_of(&As[cur ^ 1][0][0]),
                  DIM, 4096, 64, 128, DIM);
      tdm_load_2d(Wtile + k0 + 64, lds_off_of(&Bs[cur ^ 1][0][0]),
                  DIM, (unsigned)ntot, 64, 128, DIM);
    }
#pragma unroll
    for (int kk = 0; kk < 64; kk += 32) {
      v16bf af[2], bfr[4];
#pragma unroll
      for (int i = 0; i < 2; ++i)
        af[i] = ld_frag(&As[cur][wm + i * 16 + ln][kk + hh * 8],
                        &As[cur][wm + i * 16 + ln][kk + 16 + hh * 8]);
#pragma unroll
      for (int j = 0; j < 4; ++j)
        bfr[j] = ld_frag(&Bs[cur][wn + j * 16 + ln][kk + hh * 8],
                         &Bs[cur][wn + j * 16 + ln][kk + 16 + hh * 8]);
#pragma unroll
      for (int i = 0; i < 2; ++i)
#pragma unroll
        for (int j = 0; j < 4; ++j)
          acc[i][j] = wmma_bf16(af[i], bfr[j], acc[i][j]);
    }
    if (more) {
      if (wave == 0) wait_tensor0();  // DMA landed
      __syncthreads();                // visible to all + done reading cur
      cur ^= 1;
    }
  }
#elif defined(HAS_ASYNC_LDS)
  __shared__ __align__(16) __bf16 As[2][128][64];
  __shared__ __align__(16) __bf16 Bs[2][128][64];

#pragma unroll
  for (int j = 0; j < 4; ++j) {
    async_cp16(A + (size_t)(m0 + lr + j * 32) * DIM + lc, &As[0][lr + j * 32][lc]);
    async_cp16(W + (size_t)(n0 + lr + j * 32) * DIM + lc, &Bs[0][lr + j * 32][lc]);
  }
  wait_async0();
  __syncthreads();

  int cur = 0;
  for (int k0 = 0; k0 < DIM; k0 += 64) {
    const bool more = (k0 + 64) < DIM;
    if (more) {
#pragma unroll
      for (int j = 0; j < 4; ++j) {
        async_cp16(A + (size_t)(m0 + lr + j * 32) * DIM + k0 + 64 + lc,
                   &As[cur ^ 1][lr + j * 32][lc]);
        async_cp16(W + (size_t)(n0 + lr + j * 32) * DIM + k0 + 64 + lc,
                   &Bs[cur ^ 1][lr + j * 32][lc]);
      }
    }
#pragma unroll
    for (int kk = 0; kk < 64; kk += 32) {
      v16bf af[2], bfr[4];
#pragma unroll
      for (int i = 0; i < 2; ++i)
        af[i] = ld_frag(&As[cur][wm + i * 16 + ln][kk + hh * 8],
                        &As[cur][wm + i * 16 + ln][kk + 16 + hh * 8]);
#pragma unroll
      for (int j = 0; j < 4; ++j)
        bfr[j] = ld_frag(&Bs[cur][wn + j * 16 + ln][kk + hh * 8],
                         &Bs[cur][wn + j * 16 + ln][kk + 16 + hh * 8]);
#pragma unroll
      for (int i = 0; i < 2; ++i)
#pragma unroll
        for (int j = 0; j < 4; ++j)
          acc[i][j] = wmma_bf16(af[i], bfr[j], acc[i][j]);
    }
    if (more) {
      wait_async0();
      __syncthreads();
      cur ^= 1;
    }
  }
#else
  __shared__ __align__(16) __bf16 As[128][64];
  __shared__ __align__(16) __bf16 Bs[128][64];
  v8bf areg[4], breg[4];
#pragma unroll
  for (int j = 0; j < 4; ++j) {
    areg[j] = *(const v8bf*)(A + (size_t)(m0 + lr + j * 32) * DIM + lc);
    breg[j] = *(const v8bf*)(W + (size_t)(n0 + lr + j * 32) * DIM + lc);
  }
#pragma unroll
  for (int j = 0; j < 4; ++j) {
    *(v8bf*)&As[lr + j * 32][lc] = areg[j];
    *(v8bf*)&Bs[lr + j * 32][lc] = breg[j];
  }
  __syncthreads();

  for (int k0 = 0; k0 < DIM; k0 += 64) {
    const bool more = (k0 + 64) < DIM;
    if (more) {
#pragma unroll
      for (int j = 0; j < 4; ++j) {
        areg[j] = *(const v8bf*)(A + (size_t)(m0 + lr + j * 32) * DIM + k0 + 64 + lc);
        breg[j] = *(const v8bf*)(W + (size_t)(n0 + lr + j * 32) * DIM + k0 + 64 + lc);
      }
    }
#pragma unroll
    for (int kk = 0; kk < 64; kk += 32) {
      v16bf af[2], bfr[4];
#pragma unroll
      for (int i = 0; i < 2; ++i)
        af[i] = ld_frag(&As[wm + i * 16 + ln][kk + hh * 8],
                        &As[wm + i * 16 + ln][kk + 16 + hh * 8]);
#pragma unroll
      for (int j = 0; j < 4; ++j)
        bfr[j] = ld_frag(&Bs[wn + j * 16 + ln][kk + hh * 8],
                         &Bs[wn + j * 16 + ln][kk + 16 + hh * 8]);
#pragma unroll
      for (int i = 0; i < 2; ++i)
#pragma unroll
        for (int j = 0; j < 4; ++j)
          acc[i][j] = wmma_bf16(af[i], bfr[j], acc[i][j]);
    }
    if (more) {
      __syncthreads();
#pragma unroll
      for (int j = 0; j < 4; ++j) {
        *(v8bf*)&As[lr + j * 32][lc] = areg[j];
        *(v8bf*)&Bs[lr + j * 32][lc] = breg[j];
      }
      __syncthreads();
    }
  }
#endif
}

// ---------------------------------------------------------------------------
// GEMM 1: qkv = x * Wqkv^T -> per-head Q (pre-scaled by 1/sqrt(D)), K, V^T
// ---------------------------------------------------------------------------
__global__ __launch_bounds__(256)
void gemm_qkv(const __bf16* __restrict__ A, const __bf16* __restrict__ W,
              __bf16* __restrict__ qb, __bf16* __restrict__ kb,
              __bf16* __restrict__ vtb) {
  const int lane = threadIdx.x & 31;
  const int wave = threadIdx.x >> 5;
  const int hh = lane >> 4, ln = lane & 15;
  const int m0 = blockIdx.y * 128, n0 = blockIdx.x * 128;
  const int wm = (wave & 3) * 32, wn = (wave >> 2) * 64;
  v8f acc[2][4] = {};
  gemm_core(A, W, m0, n0, 6144, acc);

  const float qscale = 0.08838834764831845f;   // 1/sqrt(128) folded into Q
#pragma unroll
  for (int i = 0; i < 2; ++i) {
#pragma unroll
    for (int j = 0; j < 4; ++j) {
      int e = n0 + wn + j * 16 + ln;
      int head = e >> 7, d = e & 127;
#pragma unroll
      for (int v = 0; v < 8; ++v) {
        int m = m0 + wm + i * 16 + v + 8 * hh;
        int b = m >> 11, s = m & 2047;
        float fv = acc[i][j][v];
        if (head < 16)
          qb[(((size_t)b * 16 + head) * S_ + s) * D_ + d] = (__bf16)(fv * qscale);
        else if (head < 32)
          kb[(((size_t)b * 16 + (head - 16)) * S_ + s) * D_ + d] = (__bf16)fv;
        else
          vtb[(((size_t)b * 16 + (head - 32)) * D_ + d) * S_ + s] = (__bf16)fv;
      }
    }
  }
}

// ---------------------------------------------------------------------------
// GEMM 2: final = attn(4096x2048) * Wout(2048x2048)^T -> fp32 d_out
// ---------------------------------------------------------------------------
__global__ __launch_bounds__(256)
void gemm_out(const __bf16* __restrict__ A, const __bf16* __restrict__ W,
              float* __restrict__ out) {
  const int lane = threadIdx.x & 31;
  const int wave = threadIdx.x >> 5;
  const int hh = lane >> 4, ln = lane & 15;
  const int m0 = blockIdx.y * 128, n0 = blockIdx.x * 128;
  const int wm = (wave & 3) * 32, wn = (wave >> 2) * 64;
  v8f acc[2][4] = {};
  gemm_core(A, W, m0, n0, 2048, acc);

#pragma unroll
  for (int i = 0; i < 2; ++i)
#pragma unroll
    for (int j = 0; j < 4; ++j) {
      int e = n0 + wn + j * 16 + ln;
#pragma unroll
      for (int v = 0; v < 8; ++v) {
        int m = m0 + wm + i * 16 + v + 8 * hh;
        out[(size_t)m * DIM + e] = acc[i][j][v];
      }
    }
}

// ---------------------------------------------------------------------------
// Flash attention: grid (S/128, H, B), 8 waves x 16 queries each.
// Per 64-key tile: 16 WMMAs (logits) + online softmax + 16 WMMAs (P * V^T).
// ---------------------------------------------------------------------------
__global__ __launch_bounds__(256)
void attn_fwd(const __bf16* __restrict__ qb, const __bf16* __restrict__ kb,
              const __bf16* __restrict__ vtb, __bf16* __restrict__ attnb) {
  __shared__ __align__(16) __bf16 Plds[8][16][64];   // per-wave P tile, 16 KB
  const int wave = threadIdx.x >> 5, lane = threadIdx.x & 31;
  const int hh = lane >> 4, ln = lane & 15;
  const int bh = blockIdx.z * H_ + blockIdx.y;
  const int q0 = blockIdx.x * 128 + wave * 16;
  const __bf16* Q  = qb  + (size_t)bh * S_ * D_;
  const __bf16* Kr = kb  + (size_t)bh * S_ * D_;
  const __bf16* Vt = vtb + (size_t)bh * D_ * S_;

  v16bf qf[4];                               // Q rows resident: 16x128 bf16
  {
    const __bf16* qrow = Q + (size_t)(q0 + ln) * D_;
#pragma unroll
    for (int c = 0; c < 4; ++c)
      qf[c] = ld_frag(qrow + c * 32 + hh * 8, qrow + c * 32 + 16 + hh * 8);
  }
  v8f acc[8] = {};
  float mrow[8], lrow[8];
#pragma unroll
  for (int v = 0; v < 8; ++v) { mrow[v] = -3.0e38f; lrow[v] = 0.f; }

  for (int kt = 0; kt < S_; kt += 64) {
    if (kt + 64 < S_) {                      // prefetch next K tile rows
#pragma unroll
      for (int st = 0; st < 4; ++st)
        __builtin_prefetch(Kr + (size_t)(kt + 64 + st * 16 + ln) * D_ + hh * 64, 0, 0);
    }
    v8f s[4] = {};
#pragma unroll
    for (int c = 0; c < 4; ++c) {            // logits: Q(16x128) * K_tile^T
#pragma unroll
      for (int st = 0; st < 4; ++st) {
        const __bf16* kp = Kr + (size_t)(kt + st * 16 + ln) * D_ + c * 32 + hh * 8;
        v16bf kf = ld_frag(kp, kp + 16);
        s[st] = wmma_bf16(qf[c], kf, s[st]);
      }
    }
    float corr[8];
#pragma unroll
    for (int v = 0; v < 8; ++v) {            // online softmax per row (64 keys)
      float a0 = s[0][v], a1 = s[1][v], a2 = s[2][v], a3 = s[3][v];
      float t = fmaxf(fmaxf(a0, a1), fmaxf(a2, a3));
#pragma unroll
      for (int msk = 1; msk < 16; msk <<= 1) t = fmaxf(t, __shfl_xor(t, msk, 32));
      float mn = fmaxf(mrow[v], t);
      corr[v] = __expf(mrow[v] - mn);
      mrow[v] = mn;
      float p0 = __expf(a0 - mn), p1 = __expf(a1 - mn);
      float p2 = __expf(a2 - mn), p3 = __expf(a3 - mn);
      float rs = (p0 + p1) + (p2 + p3);
#pragma unroll
      for (int msk = 1; msk < 16; msk <<= 1) rs += __shfl_xor(rs, msk, 32);
      lrow[v] = lrow[v] * corr[v] + rs;
      const int r = v + 8 * hh;              // C-layout -> row-major P
      Plds[wave][r][ln]      = (__bf16)p0;
      Plds[wave][r][16 + ln] = (__bf16)p1;
      Plds[wave][r][32 + ln] = (__bf16)p2;
      Plds[wave][r][48 + ln] = (__bf16)p3;
    }
#pragma unroll
    for (int dt = 0; dt < 8; ++dt)
#pragma unroll
      for (int v = 0; v < 8; ++v) acc[dt][v] *= corr[v];
    asm volatile("s_wait_dscnt 0" ::: "memory");      // P stores -> A-frag loads
#pragma unroll
    for (int half = 0; half < 2; ++half) {   // out += P(16x64) * Vt(64x128)
      v16bf pf = ld_frag(&Plds[wave][ln][half * 32 + hh * 8],
                         &Plds[wave][ln][half * 32 + 16 + hh * 8]);
#pragma unroll
      for (int dt = 0; dt < 8; ++dt) {
        const __bf16* vrow = Vt + (size_t)(dt * 16 + ln) * S_ + kt + half * 32 + hh * 8;
        if (kt + 64 < S_) __builtin_prefetch(vrow + 64, 0, 0);
        v16bf vf = ld_frag(vrow, vrow + 16);
        acc[dt] = wmma_bf16(pf, vf, acc[dt]);
      }
    }
  }

  const int b = bh >> 4, head = bh & 15;
  float inv[8];
#pragma unroll
  for (int v = 0; v < 8; ++v) inv[v] = 1.0f / lrow[v];
#pragma unroll
  for (int dt = 0; dt < 8; ++dt) {
    int d = dt * 16 + ln;
#pragma unroll
    for (int v = 0; v < 8; ++v) {
      int s = q0 + v + 8 * hh;
      attnb[(size_t)(b * S_ + s) * (H_ * D_) + head * D_ + d] =
          (__bf16)(acc[dt][v] * inv[v]);
    }
  }
}

// ---------------------------------------------------------------------------
// Host launch
// ---------------------------------------------------------------------------
extern "C" void kernel_launch(void* const* d_in, const int* in_sizes, int n_in,
                              void* d_out, int out_size, void* d_ws, size_t ws_size,
                              hipStream_t stream) {
  (void)in_sizes; (void)n_in; (void)out_size; (void)ws_size;
  const float* x    = (const float*)d_in[0];
  const float* Wqkv = (const float*)d_in[1];
  const float* Wout = (const float*)d_in[2];
  char* ws = (char*)d_ws;

  __bf16* xb    = (__bf16*)(ws);
  __bf16* wqkvb = (__bf16*)(ws + (size_t)16 * 1024 * 1024);
  __bf16* woutb = (__bf16*)(ws + (size_t)40 * 1024 * 1024);
  __bf16* qb    = (__bf16*)(ws + (size_t)48 * 1024 * 1024);
  __bf16* kb    = (__bf16*)(ws + (size_t)64 * 1024 * 1024);
  __bf16* vtb   = (__bf16*)(ws + (size_t)80 * 1024 * 1024);
  __bf16* attnb = (__bf16*)(ws + (size_t)96 * 1024 * 1024);

  const int n_x = B_ * S_ * DIM / 4;
  const int n_w = 3 * H_ * D_ * DIM / 4;
  const int n_o = DIM * H_ * D_ / 4;
  cvt_f32_bf16<<<n_x / 256, 256, 0, stream>>>(x, xb, n_x);
  cvt_f32_bf16<<<n_w / 256, 256, 0, stream>>>(Wqkv, wqkvb, n_w);
  cvt_f32_bf16<<<n_o / 256, 256, 0, stream>>>(Wout, woutb, n_o);

  gemm_qkv<<<dim3(48, 32), 256, 0, stream>>>(xb, wqkvb, qb, kb, vtb);
  attn_fwd<<<dim3(S_ / 128, H_, B_), 256, 0, stream>>>(qb, kb, vtb, attnb);
  gemm_out<<<dim3(16, 32), 256, 0, stream>>>(attnb, woutb, (float*)d_out);
}
